// GNN_33200097198207
// MI455X (gfx1250) — matrix operand order repaired and checked by reading, not compiled, not run on hardware
//
#include <hip/hip_runtime.h>

#define HD   128
#define NN   50000
#define EE   800000
#define ROWH 136      // LDS row stride in halves (128 + 8 pad)
#define TILE 64       // rows per workgroup tile

typedef __attribute__((ext_vector_type(16))) __bf16       v16bf;
typedef __attribute__((ext_vector_type(8)))  float        v8f;
typedef __attribute__((ext_vector_type(4)))  unsigned int v4u;

struct B32frag { v4u lo, hi; };

static __device__ __forceinline__ v16bf frag_of(v4u lo, v4u hi) {
    B32frag t; t.lo = lo; t.hi = hi;
    return __builtin_bit_cast(v16bf, t);
}

static __device__ __forceinline__ unsigned short f2bf(float x) {
    unsigned int u = __builtin_bit_cast(unsigned int, x);
    return (unsigned short)((u + 0x7fffu + ((u >> 16) & 1u)) >> 16);
}

static __device__ __forceinline__ float silu_f(float x) {
    return x / (1.0f + __expf(-x));
}

static __device__ __forceinline__ v8f zero8() {
    v8f z = {0.f, 0.f, 0.f, 0.f, 0.f, 0.f, 0.f, 0.f};
    return z;
}

// A fragment (16x32 bf16, MxK) from an LDS tile (row-major, ROWH stride).
// ISA layout: lanes 0-15 hold row M=lane, K=0..7 & 16..23; lanes 16-31 hold
// row M=lane-16, K=8..15 & 24..31 -> two 16B contiguous chunks 32B apart.
static __device__ __forceinline__ v16bf ldA(const unsigned short* tile, int lane, int kt) {
    int m  = lane & 15;
    int hs = lane >> 4;
    const v4u* p = (const v4u*)(tile + m * ROWH + kt * 32 + hs * 8);
    return frag_of(p[0], p[2]);
}

// B fragment (32x16 bf16, KxN) from fragment-packed global weights:
// per (kt,nt) fragment, lane holds 16 contiguous halves (32B).
static __device__ __forceinline__ v16bf ldB(const unsigned short* pack, int lane, int kt, int nt) {
    const v4u* p = (const v4u*)pack + ((((kt * 8 + nt) * 32) + lane) << 1);
    return frag_of(p[0], p[1]);
}

#define WMMA_BF16(a, b, c) \
    __builtin_amdgcn_wmma_f32_16x16x32_bf16(false, (a), false, (b), (short)0, (c), false, false)

// ---------------------------------------------------------------------------
// Repack f32 weights [128x128] -> bf16 WMMA fragment order.
// Packed element p = ((mat*4 + kt)*8 + nt)*32*16 + lane*16 + e, where element
// e (0..15) of lane L maps to W[kt*32 + (L/16)*8 + (e%8) + (e/8)*16][nt*16 + L%16].
// ---------------------------------------------------------------------------
__global__ void gnn_repack(const float* __restrict__ Wu,  const float* __restrict__ Wv,
                           const float* __restrict__ eW2, const float* __restrict__ eW3,
                           const float* __restrict__ Wh,  const float* __restrict__ Wa1,
                           const float* __restrict__ Wa2, const float* __restrict__ uW2,
                           unsigned short* __restrict__ pack) {
    int p = blockIdx.x * blockDim.x + threadIdx.x;
    if (p >= 24 * 16384) return;
    int mat  = p >> 14;
    int q    = p & 16383;
    int e    = q & 15;
    int lane = (q >> 4) & 31;
    int nt   = (q >> 9) & 7;
    int kt   = q >> 12;
    int k    = kt * 32 + (lane >> 4) * 8 + (e & 7) + (e >> 3) * 16;
    int n    = nt * 16 + (lane & 15);
    const float* src;
    if      (mat <  4) src = Wu  + (size_t)(mat     ) * HD * HD;
    else if (mat <  8) src = Wv  + (size_t)(mat -  4) * HD * HD;
    else if (mat < 12) src = eW2 + (size_t)(mat -  8) * HD * HD;
    else if (mat < 16) src = eW3 + (size_t)(mat - 12) * HD * HD;
    else if (mat < 18) src = Wh  + (size_t)(mat - 16) * HD * HD;
    else if (mat < 20) src = Wa1 + (size_t)(mat - 18) * HD * HD;
    else if (mat < 22) src = Wa2 + (size_t)(mat - 20) * HD * HD;
    else               src = uW2 + (size_t)(mat - 22) * HD * HD;
    pack[p] = f2bf(src[k * HD + n]);
}

// ---------------------------------------------------------------------------
// Embedding gather: h[n,:] = emb[atom_num[n],:]  (f32 + bf16 copies)
// ---------------------------------------------------------------------------
__global__ void gnn_embed(const int* __restrict__ atom, const float* __restrict__ emb,
                          float* __restrict__ hf, unsigned short* __restrict__ hbf) {
    int i = blockIdx.x * blockDim.x + threadIdx.x;
    if (i >= NN * HD) return;
    int r = i >> 7, c = i & 127;
    float v = emb[atom[r] * HD + c];
    hf[i]  = v;
    hbf[i] = f2bf(v);
}

// ---------------------------------------------------------------------------
// Edge block: m = (silu(silu(h[u]Wu + h[v]Wv + dis*Wdis + b1) W2 + b2)) W3 + b3
// fused with segment-sum scatter: aout[u] += m ; aout[v] += m.
// 128 threads = 4 waves, 64 edges per workgroup, 16 edges per wave.
// A fragments are held in registers; nt is the outer loop so only one 8-VGPR
// accumulator is live -> scheduler can pipeline B-fragment global loads.
// ---------------------------------------------------------------------------
__global__ __launch_bounds__(128) void gnn_edge_block(
    const unsigned short* __restrict__ hbf,
    const int*   __restrict__ idu,  const int*   __restrict__ idv,
    const float* __restrict__ dis,
    const unsigned short* __restrict__ pWu, const unsigned short* __restrict__ pWv,
    const unsigned short* __restrict__ pW2, const unsigned short* __restrict__ pW3,
    const float* __restrict__ wdis, const float* __restrict__ b1,
    const float* __restrict__ b2,   const float* __restrict__ b3,
    float* __restrict__ aout) {

    __shared__ __align__(16) unsigned short sU[TILE * ROWH];
    __shared__ __align__(16) unsigned short sV[TILE * ROWH];
    __shared__ int   sIu[TILE];
    __shared__ int   sIv[TILE];
    __shared__ float sD [TILE];

    const int tid  = threadIdx.x;
    const int wave = tid >> 5;
    const int lane = tid & 31;
    const int e0   = blockIdx.x * TILE;

    if (tid < TILE) {
        sIu[tid] = idu[e0 + tid];
        sIv[tid] = idv[e0 + tid];
        sD [tid] = dis[e0 + tid];
    }
    __syncthreads();

    // Stage endpoint feature rows (bf16) into LDS: 64 rows x 8 x 16B.
    for (int i = tid; i < TILE * 8; i += 128) {
        int r = i >> 3, q = i & 7;
        ((v4u*)(sU + r * ROWH))[q] = ((const v4u*)(hbf + (size_t)sIu[r] * HD))[q];
        ((v4u*)(sV + r * ROWH))[q] = ((const v4u*)(hbf + (size_t)sIv[r] * HD))[q];
    }
    __syncthreads();

    unsigned short* myU = sU + wave * 16 * ROWH;   // this wave's 16 rows
    unsigned short* myV = sV + wave * 16 * ROWH;
    const int nb = lane & 15;
    const int mh = (lane >> 4) << 3;

    // Per-row scalars for this lane's 8 output rows (hoisted out of nt loops).
    float dv[8];
    int   uu[8], vv[8];
    #pragma unroll
    for (int r = 0; r < 8; ++r) {
        int row = wave * 16 + mh + r;
        dv[r] = sD[row];
        uu[r] = sIu[row];
        vv[r] = sIv[row];
    }

    // ---- GEMM 1: h[u]@Wu + h[v]@Wv ----
    v16bf au[4], av[4];
    #pragma unroll
    for (int kt = 0; kt < 4; ++kt) {
        au[kt] = ldA(myU, lane, kt);
        av[kt] = ldA(myV, lane, kt);
    }
    #pragma unroll
    for (int nt = 0; nt < 8; ++nt) {
        v8f acc = zero8();
        #pragma unroll
        for (int kt = 0; kt < 4; ++kt) {
            acc = WMMA_BF16(au[kt], ldB(pWu, lane, kt, nt), acc);
            acc = WMMA_BF16(av[kt], ldB(pWv, lane, kt, nt), acc);
        }
        int n = nt * 16 + nb;
        float wd = wdis[n], bb = b1[n];
        #pragma unroll
        for (int r = 0; r < 8; ++r) {
            int m = mh + r;
            myU[m * ROWH + n] = f2bf(silu_f(acc[r] + dv[r] * wd + bb));
        }
    }

    // ---- GEMM 2: t @ W2 ----  (A = t from myU; per-wave LDS is in order)
    v16bf at[4];
    #pragma unroll
    for (int kt = 0; kt < 4; ++kt) at[kt] = ldA(myU, lane, kt);
    #pragma unroll
    for (int nt = 0; nt < 8; ++nt) {
        v8f acc = zero8();
        #pragma unroll
        for (int kt = 0; kt < 4; ++kt)
            acc = WMMA_BF16(at[kt], ldB(pW2, lane, kt, nt), acc);
        int n = nt * 16 + nb;
        float bb = b2[n];
        #pragma unroll
        for (int r = 0; r < 8; ++r) {
            int m = mh + r;
            myV[m * ROWH + n] = f2bf(silu_f(acc[r] + bb));
        }
    }

    // ---- GEMM 3: t2 @ W3 + b3, fused atomic scatter into aout ----
    v16bf a3[4];
    #pragma unroll
    for (int kt = 0; kt < 4; ++kt) a3[kt] = ldA(myV, lane, kt);
    #pragma unroll
    for (int nt = 0; nt < 8; ++nt) {
        v8f acc = zero8();
        #pragma unroll
        for (int kt = 0; kt < 4; ++kt)
            acc = WMMA_BF16(a3[kt], ldB(pW3, lane, kt, nt), acc);
        int n = nt * 16 + nb;
        float bb = b3[n];
        #pragma unroll
        for (int r = 0; r < 8; ++r) {
            float x = acc[r] + bb;
            unsafeAtomicAdd(aout + (size_t)uu[r] * HD + n, x);
            unsafeAtomicAdd(aout + (size_t)vv[r] * HD + n, x);
        }
    }
}

// ---------------------------------------------------------------------------
// Atom update: t = silu(h@Wh + a1@Wa1 + a2@Wa2 + b1); h += t@W2 + b2
// ---------------------------------------------------------------------------
__global__ __launch_bounds__(128) void gnn_atom_update(
    float* __restrict__ hf, unsigned short* __restrict__ hbf,
    const float* __restrict__ a1, const float* __restrict__ a2,
    const unsigned short* __restrict__ pWh,  const unsigned short* __restrict__ pWa1,
    const unsigned short* __restrict__ pWa2, const unsigned short* __restrict__ pW2,
    const float* __restrict__ b1, const float* __restrict__ b2) {

    __shared__ __align__(16) unsigned short sH [TILE * ROWH];
    __shared__ __align__(16) unsigned short sM1[TILE * ROWH];
    __shared__ __align__(16) unsigned short sM2[TILE * ROWH];

    const int tid  = threadIdx.x;
    const int wave = tid >> 5;
    const int lane = tid & 31;
    const int row0 = blockIdx.x * TILE;

    for (int i = tid; i < TILE * HD; i += 128) {
        int r = i >> 7, c = i & 127;
        int rg = row0 + r; if (rg > NN - 1) rg = NN - 1;
        size_t off = (size_t)rg * HD + c;
        sH [r * ROWH + c] = hbf[off];
        sM1[r * ROWH + c] = f2bf(a1[off]);
        sM2[r * ROWH + c] = f2bf(a2[off]);
    }
    __syncthreads();

    unsigned short* myH        = sH  + wave * 16 * ROWH;
    const unsigned short* myM1 = sM1 + wave * 16 * ROWH;
    const unsigned short* myM2 = sM2 + wave * 16 * ROWH;
    const int nb = lane & 15;
    const int mh = (lane >> 4) << 3;

    // ---- GEMM 1: h@Wh + a1@Wa1 + a2@Wa2 ----
    v16bf ah[4], am1[4], am2[4];
    #pragma unroll
    for (int kt = 0; kt < 4; ++kt) {
        ah [kt] = ldA(myH,  lane, kt);
        am1[kt] = ldA(myM1, lane, kt);
        am2[kt] = ldA(myM2, lane, kt);
    }
    #pragma unroll
    for (int nt = 0; nt < 8; ++nt) {
        v8f acc = zero8();
        #pragma unroll
        for (int kt = 0; kt < 4; ++kt) {
            acc = WMMA_BF16(ah [kt], ldB(pWh,  lane, kt, nt), acc);
            acc = WMMA_BF16(am1[kt], ldB(pWa1, lane, kt, nt), acc);
            acc = WMMA_BF16(am2[kt], ldB(pWa2, lane, kt, nt), acc);
        }
        int n = nt * 16 + nb;
        float bb = b1[n];
        #pragma unroll
        for (int r = 0; r < 8; ++r) {
            int m = mh + r;
            myH[m * ROWH + n] = f2bf(silu_f(acc[r] + bb));
        }
    }

    // ---- GEMM 2: t @ W2, residual, write back f32 + bf16 ----
    v16bf at[4];
    #pragma unroll
    for (int kt = 0; kt < 4; ++kt) at[kt] = ldA(myH, lane, kt);
    #pragma unroll
    for (int nt = 0; nt < 8; ++nt) {
        v8f acc = zero8();
        #pragma unroll
        for (int kt = 0; kt < 4; ++kt)
            acc = WMMA_BF16(at[kt], ldB(pW2, lane, kt, nt), acc);
        int n = nt * 16 + nb;
        float bb = b2[n];
        #pragma unroll
        for (int r = 0; r < 8; ++r) {
            int rg = row0 + wave * 16 + mh + r;
            if (rg < NN) {
                size_t off = (size_t)rg * HD + n;
                float hn = hf[off] + acc[r] + bb;
                hf[off]  = hn;
                hbf[off] = f2bf(hn);
            }
        }
    }
}

// ---------------------------------------------------------------------------
// Output head: out = h @ outW + outb   (outW: [128,3])
// ---------------------------------------------------------------------------
__global__ void gnn_head(const float* __restrict__ hf, const float* __restrict__ outW,
                         const float* __restrict__ outb, float* __restrict__ out) {
    int i = blockIdx.x * blockDim.x + threadIdx.x;
    if (i >= NN) return;
    const float* row = hf + (size_t)i * HD;
    float a0 = outb[0], a1 = outb[1], a2 = outb[2];
    #pragma unroll 4
    for (int k = 0; k < HD; ++k) {
        float x = row[k];
        a0 += x * outW[k * 3 + 0];
        a1 += x * outW[k * 3 + 1];
        a2 += x * outW[k * 3 + 2];
    }
    out[i * 3 + 0] = a0;
    out[i * 3 + 1] = a1;
    out[i * 3 + 2] = a2;
}

// ---------------------------------------------------------------------------
extern "C" void kernel_launch(void* const* d_in, const int* in_sizes, int n_in,
                              void* d_out, int out_size, void* d_ws, size_t ws_size,
                              hipStream_t stream) {
    (void)in_sizes; (void)n_in; (void)out_size; (void)ws_size;

    const int*   atom_num = (const int*)  d_in[0];
    const float* dis1     = (const float*)d_in[1];
    const float* dis2     = (const float*)d_in[2];
    const int*   id1u     = (const int*)  d_in[3];
    const int*   id1v     = (const int*)  d_in[4];
    const int*   id2u     = (const int*)  d_in[5];
    const int*   id2v     = (const int*)  d_in[6];
    const float* emb      = (const float*)d_in[7];
    const float* Wu       = (const float*)d_in[8];
    const float* Wv       = (const float*)d_in[9];
    const float* Wdis     = (const float*)d_in[10];
    const float* eb1      = (const float*)d_in[11];
    const float* eW2      = (const float*)d_in[12];
    const float* eb2      = (const float*)d_in[13];
    const float* eW3      = (const float*)d_in[14];
    const float* eb3      = (const float*)d_in[15];
    const float* Wh       = (const float*)d_in[16];
    const float* Wa1      = (const float*)d_in[17];
    const float* Wa2      = (const float*)d_in[18];
    const float* ub1      = (const float*)d_in[19];
    const float* uW2      = (const float*)d_in[20];
    const float* ub2      = (const float*)d_in[21];
    const float* outW     = (const float*)d_in[22];
    const float* outb     = (const float*)d_in[23];

    char* ws = (char*)d_ws;
    const size_t szHf  = (size_t)NN * HD * sizeof(float);          // 25.6 MB
    const size_t szHbf = (size_t)NN * HD * sizeof(unsigned short); // 12.8 MB
    float*          hf   = (float*)(ws);
    unsigned short* hbf  = (unsigned short*)(ws + szHf);
    float*          agg1 = (float*)(ws + szHf + szHbf);
    float*          agg2 = (float*)(ws + 2 * szHf + szHbf);
    unsigned short* pack = (unsigned short*)(ws + 3 * szHf + szHbf);

    // Packed weight matrix offsets (in ushort units of 16384 per matrix):
    // 0..3: Wu[b], 4..7: Wv[b], 8..11: eW2[b], 12..15: eW3[b],
    // 16..17: Wh[j], 18..19: Wa1[j], 20..21: Wa2[j], 22..23: uW2[j]
    auto pm = [&](int idx) { return pack + (size_t)idx * 16384; };

    gnn_repack<<<(24 * 16384 + 255) / 256, 256, 0, stream>>>(
        Wu, Wv, eW2, eW3, Wh, Wa1, Wa2, uW2, pack);
    gnn_embed<<<(NN * HD + 255) / 256, 256, 0, stream>>>(atom_num, emb, hf, hbf);

    for (int round = 0; round < 2; ++round) {
        hipMemsetAsync(agg1, 0, szHf, stream);
        hipMemsetAsync(agg2, 0, szHf, stream);
        int b0 = round * 2;       // edge blocks (b0, b0+1)
        gnn_edge_block<<<EE / TILE, 128, 0, stream>>>(
            hbf, id1u, id1v, dis1,
            pm(0 + b0), pm(4 + b0), pm(8 + b0), pm(12 + b0),
            Wdis + (size_t)b0 * HD, eb1 + (size_t)b0 * HD,
            eb2 + (size_t)b0 * HD, eb3 + (size_t)b0 * HD, agg1);
        gnn_edge_block<<<EE / TILE, 128, 0, stream>>>(
            hbf, id2u, id2v, dis2,
            pm(1 + b0), pm(5 + b0), pm(9 + b0), pm(13 + b0),
            Wdis + (size_t)(b0 + 1) * HD, eb1 + (size_t)(b0 + 1) * HD,
            eb2 + (size_t)(b0 + 1) * HD, eb3 + (size_t)(b0 + 1) * HD, agg2);
        int j = round;            // update block
        gnn_atom_update<<<(NN + TILE - 1) / TILE, 128, 0, stream>>>(
            hf, hbf, agg1, agg2,
            pm(16 + j), pm(18 + j), pm(20 + j), pm(22 + j),
            ub1 + (size_t)j * HD, ub2 + (size_t)j * HD);
    }

    gnn_head<<<(NN + 255) / 256, 256, 0, stream>>>(hf, outW, outb, (float*)d_out);
}
